// Enc_Dec_38250978738655
// MI455X (gfx1250) — matrix operand order repaired and checked by reading, not compiled, not run on hardware
//
#include <hip/hip_runtime.h>
#include <hip/hip_bf16.h>

typedef __attribute__((ext_vector_type(16))) _Float16 v16h;
typedef __attribute__((ext_vector_type(8)))  _Float16 v8h;
typedef __attribute__((ext_vector_type(8)))  float    v8f;

// ---------------- model constants ----------------
enum { B_ = 4, P_ = 8, T_ = 8, N_ = 500, E_ = 2000, F_ = 64, DIN = 128 };
enum { NP = 512, EP = 2048 };   // padded node / edge counts (mult of 16/32)

// ---- branch-free transcendentals on the TRANS pipe (v_exp_f32 / v_rcp_f32) ----
#define LOG2E 1.4426950408889634f
__device__ __forceinline__ float fast_exp(float x) {           // x <= 0 in our uses
    return __builtin_amdgcn_exp2f(x * LOG2E);
}
__device__ __forceinline__ float fast_sigmoid(float x) {
    float t = __builtin_amdgcn_exp2f(fminf(-x * LOG2E, 88.f)); // avoid inf
    return __builtin_amdgcn_rcpf(1.f + t);
}
__device__ __forceinline__ float fast_tanh(float x) {
    float ax = fminf(fabsf(x), 15.f);
    float t  = __builtin_amdgcn_exp2f(ax * (-2.f * LOG2E));    // e^(-2|x|)
    float r  = (1.f - t) * __builtin_amdgcn_rcpf(1.f + t);
    return x < 0.f ? -r : r;
}

// =====================================================================
// Generic batched WMMA GEMM:  C[b] = act( A[b] (MxK,f16) * B[b] (KxN,f16) + bias )
// M % 128 == 0, N % 64 == 0, K % 64 == 0 (guaranteed by padding).
// Per block: 128x64 tile, 8 waves, each wave a 32x32 sub-tile (4 wmma accums).
// K-loop unrolled by two 32-chunks per iteration.
// act: 0 none, 1 sigmoid, 2 tanh
// =====================================================================
__global__ __launch_bounds__(256) void gemm_wmma_kernel(
    const _Float16* __restrict__ A, unsigned long long aBatch, int lda,
    const _Float16* __restrict__ Bm, unsigned long long bBatch, int ldb,
    float* __restrict__ Cf, unsigned long long cfBatch, int ldcf,
    _Float16* __restrict__ Ch, unsigned long long chBatch, int ldch,
    const float* __restrict__ bias, int Kd, int act)
{
    __shared__ __align__(16) _Float16 As[128 * 32];   // [row][k]
    __shared__ __align__(16) _Float16 Bs[64 * 32];    // [col][k] (transposed)

    const int tid  = threadIdx.x;
    const int lane = tid & 31;
    const int wave = tid >> 5;
    const int wm   = wave & 3;          // 4 row-groups of 32
    const int wn   = wave >> 2;         // 2 col-groups of 32
    const int rowBase = blockIdx.y * 128;
    const int colBase = blockIdx.x * 64;

    const _Float16* Ag = A  + (size_t)blockIdx.z * aBatch + (size_t)rowBase * lda;
    const _Float16* Bg = Bm + (size_t)blockIdx.z * bBatch + colBase;

    v8f acc[2][2] = {};

    const int rsel = lane & 15;
    const int hi   = lane >> 4;

    auto chunk = [&](int k0) {
        __syncthreads();
        // stage A tile: 128 rows x 32 halves, 16B vector loads
        #pragma unroll
        for (int i = 0; i < 2; ++i) {
            int idx = tid + i * 256;        // 0..511
            int r   = idx >> 2;             // 0..127
            int kc  = (idx & 3) << 3;       // 0,8,16,24
            *(v8h*)&As[r * 32 + kc] = *(const v8h*)&Ag[(size_t)r * lda + k0 + kc];
        }
        // stage B tile transposed: read (k, n..n+7), scatter to Bs[n][k]
        {
            int k  = tid >> 3;              // 0..31
            int n8 = (tid & 7) << 3;        // 0..56
            v8h bv = *(const v8h*)&Bg[(size_t)(k0 + k) * ldb + n8];
            #pragma unroll
            for (int j = 0; j < 8; ++j) Bs[(n8 + j) * 32 + k] = bv[j];
        }
        __syncthreads();

        // A fragment (16x32 f16): lanes 0-15 -> K 0..7 / 16..23, lanes 16-31 -> K 8..15 / 24..31
        v16h af[2], bf[2];
        #pragma unroll
        for (int tm = 0; tm < 2; ++tm) {
            int ar = (wm * 32 + tm * 16 + rsel) * 32;
            v8h lo = *(const v8h*)&As[ar + hi * 8];
            v8h hw = *(const v8h*)&As[ar + 16 + hi * 8];
            af[tm] = __builtin_shufflevector(lo, hw, 0,1,2,3,4,5,6,7,8,9,10,11,12,13,14,15);
        }
        // B fragment (32x16 f16): lanes 0-15 -> K 0..7 / 16..23, lanes 16-31 -> K 8..15 / 24..31
        #pragma unroll
        for (int tn = 0; tn < 2; ++tn) {
            int bc = (wn * 32 + tn * 16 + rsel) * 32 + hi * 8;
            v8h lo = *(const v8h*)&Bs[bc];
            v8h hw = *(const v8h*)&Bs[bc + 16];
            bf[tn] = __builtin_shufflevector(lo, hw, 0,1,2,3,4,5,6,7,8,9,10,11,12,13,14,15);
        }
        #pragma unroll
        for (int tm = 0; tm < 2; ++tm)
            #pragma unroll
            for (int tn = 0; tn < 2; ++tn)
                acc[tm][tn] = __builtin_amdgcn_wmma_f32_16x16x32_f16(
                    false, af[tm], false, bf[tn], (short)0, acc[tm][tn], false, false);
    };

    for (int k0 = 0; k0 < Kd; k0 += 64) {   // K % 64 == 0 by construction
        chunk(k0);
        chunk(k0 + 32);
    }

    // epilogue: C/D layout -> lane 0-15: N=lane, M=r(0..7); lane 16-31: N=lane-16, M=8+r
    const size_t cfOff = (size_t)blockIdx.z * cfBatch;
    const size_t chOff = (size_t)blockIdx.z * chBatch;
    #pragma unroll
    for (int tm = 0; tm < 2; ++tm)
        #pragma unroll
        for (int tn = 0; tn < 2; ++tn) {
            int col  = colBase + wn * 32 + tn * 16 + rsel;
            int row0 = rowBase + wm * 32 + tm * 16 + hi * 8;
            float bv = bias ? bias[col] : 0.f;
            #pragma unroll
            for (int r = 0; r < 8; ++r) {
                float v = acc[tm][tn][r] + bv;
                if (act == 1)      v = fast_sigmoid(v);
                else if (act == 2) v = fast_tanh(v);
                if (Cf) Cf[cfOff + (size_t)(row0 + r) * ldcf + col] = v;
                if (Ch) Ch[chOff + (size_t)(row0 + r) * ldch + col] = (_Float16)v;
            }
        }
}

// ---------------- prep kernels ----------------
// row-normalize (sum over cols) f32 -> padded f16 (zero pad both dims)
__global__ void rownorm_kernel(const float* __restrict__ src, _Float16* __restrict__ dst,
                               int rows, int cols, int colsPad)
{
    __shared__ float red[256];
    int row = blockIdx.x;
    float s = 0.f;
    if (row < rows)
        for (int c = threadIdx.x; c < cols; c += 256) s += src[(size_t)row * cols + c];
    red[threadIdx.x] = s; __syncthreads();
    for (int off = 128; off > 0; off >>= 1) {
        if (threadIdx.x < off) red[threadIdx.x] += red[threadIdx.x + off];
        __syncthreads();
    }
    float inv = 1.f / (red[0] + 1e-6f);
    for (int c = threadIdx.x; c < colsPad; c += 256) {
        float v = (row < rows && c < cols) ? src[(size_t)row * cols + c] * inv : 0.f;
        dst[(size_t)row * colsPad + c] = (_Float16)v;
    }
}

// column-normalize M (N x E) and store transposed padded: McT[e][n] (EP x NP)
__global__ void colnormT_kernel(const float* __restrict__ src, _Float16* __restrict__ dstT,
                                int rows, int cols, int rowsPadOut)
{
    __shared__ float red[256];
    int e = blockIdx.x;
    float s = 0.f;
    if (e < cols)
        for (int n = threadIdx.x; n < rows; n += 256) s += src[(size_t)n * cols + e];
    red[threadIdx.x] = s; __syncthreads();
    for (int off = 128; off > 0; off >>= 1) {
        if (threadIdx.x < off) red[threadIdx.x] += red[threadIdx.x + off];
        __syncthreads();
    }
    float inv = 1.f / (red[0] + 1e-6f);
    for (int n = threadIdx.x; n < rowsPadOut; n += 256) {
        float v = (e < cols && n < rows) ? src[(size_t)n * cols + e] * inv : 0.f;
        dstT[(size_t)e * rowsPadOut + n] = (_Float16)v;
    }
}

__global__ void f32_to_f16_kernel(const float* __restrict__ src, _Float16* __restrict__ dst, int n)
{
    int i = blockIdx.x * blockDim.x + threadIdx.x;
    if (i < n) dst[i] = (_Float16)src[i];
}

// enc_in[b,p,n,f] = x[b,p,n] * Wi[f] + bi[f]
__global__ void enc_in_kernel(const float* __restrict__ x, const float* __restrict__ Wi,
                              const float* __restrict__ bi, float* __restrict__ out, int total)
{
    int i = blockIdx.x * blockDim.x + threadIdx.x;
    if (i >= total) return;
    int f = i & 63;
    out[i] = x[i >> 6] * Wi[f] + bi[f];
}

// concat pack: h0[b,row,0:64] = xt ; h0[b,row,64:128] = (gate?)*hx ; pad rows -> 0
__global__ void pack_kernel(const float* __restrict__ xt, unsigned long long xtBatch, int xtRow,
                            const float* __restrict__ hx, const float* __restrict__ gates,
                            _Float16* __restrict__ out, int total)
{
    int i = blockIdx.x * blockDim.x + threadIdx.x;
    if (i >= total) return;
    int d = i & 127;
    int row = (i >> 7) & (NP - 1);
    int b = i >> 16;                       // NP*DIN == 65536
    float v = 0.f;
    if (row < N_) {
        if (d < F_) {
            v = xt[(size_t)b * xtBatch + (size_t)row * xtRow + d];
        } else {
            float h = hx[((size_t)b * NP + row) * F_ + (d - F_)];
            if (gates) h *= gates[((size_t)b * NP + row) * DIN + (d - F_)];
            v = h;
        }
    }
    out[i] = (_Float16)v;
}

// K=2 multi-range attention: softmax over tanh-scores, combine, store f16
__global__ void attn_kernel(const float* __restrict__ H1, const float* __restrict__ H2,
                            const float* __restrict__ avec, _Float16* __restrict__ out,
                            int ldOut, int rowsTotal)
{
    int i = blockIdx.x * blockDim.x + threadIdx.x;
    if (i >= rowsTotal) return;
    const float* h1 = H1 + (size_t)i * F_;
    const float* h2 = H2 + (size_t)i * F_;
    float s1 = 0.f, s2 = 0.f;
    for (int d = 0; d < F_; ++d) {
        float a = avec[d];
        s1 += fast_tanh(h1[d]) * a;
        s2 += fast_tanh(h2[d]) * a;
    }
    float m = fmaxf(s1, s2);
    float e1 = fast_exp(s1 - m), e2 = fast_exp(s2 - m);
    float w1 = e1 * __builtin_amdgcn_rcpf(e1 + e2), w2 = 1.f - w1;
    _Float16* o = out + (size_t)i * ldOut;
    for (int d = 0; d < F_; ++d) o[d] = (_Float16)(w1 * h1[d] + w2 * h2[d]);
}

// GRU combine: h = u*h + (1-u)*c   (gates already sigmoid, c already tanh)
__global__ void gru_kernel(const float* __restrict__ gates, const float* __restrict__ c,
                           float* __restrict__ hx, _Float16* __restrict__ hx16, int total)
{
    int i = blockIdx.x * blockDim.x + threadIdx.x;
    if (i >= total) return;
    int d = i & 63;
    int rb = i >> 6;                       // b*NP + row
    float u = gates[(size_t)rb * DIN + F_ + d];
    float hn = u * hx[i] + (1.f - u) * c[i];
    hx[i] = hn;
    hx16[i] = (_Float16)hn;
}

// final: out[b,t,n] = dot(yt[b,n,:], Wout) + bout
__global__ void final_out_kernel(const float* __restrict__ yt, const float* __restrict__ Wout,
                                 const float* __restrict__ bout, float* __restrict__ out, int t)
{
    int i = blockIdx.x * blockDim.x + threadIdx.x;
    if (i >= B_ * N_) return;
    int b = i / N_, n = i - b * N_;
    const float* y = yt + ((size_t)b * NP + n) * F_;
    float s = bout[0];
    for (int d = 0; d < F_; ++d) s += y[d] * Wout[d];
    out[((size_t)b * T_ + t) * N_ + n] = s;
}

// ---------------- host orchestration ----------------
namespace {

struct MraP {
    const _Float16 *Wn0, *Wn1, *We0, *We1, *Wo;
    const float *an, *ae, *bo;
    int dout;
};

struct Ws {
    _Float16 *An16, *Ae16, *Mr16, *McT16;
    float *enc_in, *hx, *enc_out, *gates, *cbuf, *yt;
    float *H1n, *H2n, *H1e, *H2e;
    _Float16 *hx16, *h0, *hnA, *hnB, *he0, *heA, *heB, *cat2, *eout;
};

inline void gemm(hipStream_t s, const _Float16* A, size_t aB, int lda,
                 const _Float16* Bm, size_t bB, int ldb,
                 float* Cf, size_t cfB, int ldcf,
                 _Float16* Ch, size_t chB, int ldch,
                 const float* bias, int M, int N, int Kd, int act)
{
    dim3 grid(N / 64, M / 128, B_);
    gemm_wmma_kernel<<<grid, 256, 0, s>>>(A, (unsigned long long)aB, lda,
                                          Bm, (unsigned long long)bB, ldb,
                                          Cf, (unsigned long long)cfB, ldcf,
                                          Ch, (unsigned long long)chB, ldch,
                                          bias, Kd, act);
}

void run_mra(hipStream_t s, Ws& w, const MraP& p, float* outF32, int act)
{
    const size_t NPD = (size_t)NP * DIN, NPF = (size_t)NP * F_;
    const size_t EPD = (size_t)EP * DIN, EPF = (size_t)EP * F_;
    // node branch: two hops + per-hop projection
    gemm(s, w.An16, 0, NP,  w.h0,  NPD, DIN, nullptr, 0, 0, w.hnA, NPD, DIN, nullptr, NP, DIN, NP, 0);
    gemm(s, w.hnA,  NPD, DIN, p.Wn0, 0, F_, w.H1n, NPF, F_, nullptr, 0, 0, nullptr, NP, F_, DIN, 0);
    gemm(s, w.An16, 0, NP,  w.hnA, NPD, DIN, nullptr, 0, 0, w.hnB, NPD, DIN, nullptr, NP, DIN, NP, 0);
    gemm(s, w.hnB,  NPD, DIN, p.Wn1, 0, F_, w.H2n, NPF, F_, nullptr, 0, 0, nullptr, NP, F_, DIN, 0);
    attn_kernel<<<(B_ * NP) / 256, 256, 0, s>>>(w.H1n, w.H2n, p.an, w.cat2, DIN, B_ * NP);
    // edge branch: lift, two hops + projection, attention, pull-back
    gemm(s, w.McT16, 0, NP, w.h0,  NPD, DIN, nullptr, 0, 0, w.he0, EPD, DIN, nullptr, EP, DIN, NP, 0);
    gemm(s, w.Ae16,  0, EP, w.he0, EPD, DIN, nullptr, 0, 0, w.heA, EPD, DIN, nullptr, EP, DIN, EP, 0);
    gemm(s, w.heA, EPD, DIN, p.We0, 0, F_, w.H1e, EPF, F_, nullptr, 0, 0, nullptr, EP, F_, DIN, 0);
    gemm(s, w.Ae16,  0, EP, w.heA, EPD, DIN, nullptr, 0, 0, w.heB, EPD, DIN, nullptr, EP, DIN, EP, 0);
    gemm(s, w.heB, EPD, DIN, p.We1, 0, F_, w.H2e, EPF, F_, nullptr, 0, 0, nullptr, EP, F_, DIN, 0);
    attn_kernel<<<(B_ * EP) / 256, 256, 0, s>>>(w.H1e, w.H2e, p.ae, w.eout, F_, B_ * EP);
    gemm(s, w.Mr16, 0, EP, w.eout, EPF, F_, nullptr, 0, 0, w.cat2 + F_, NPD, DIN, nullptr, NP, F_, EP, 0);
    // output projection (fused bias + activation)
    gemm(s, w.cat2, NPD, DIN, p.Wo, 0, p.dout, outF32, (size_t)NP * p.dout, p.dout,
         nullptr, 0, 0, p.bo, NP, p.dout, DIN, act);
}

void run_cell(hipStream_t s, Ws& w, const float* xt, size_t xtBatch,
              const MraP& gate, const MraP& upd)
{
    const int totP = B_ * NP * DIN;      // 262144
    const int totH = B_ * NP * F_;       // 131072
    pack_kernel<<<totP / 256, 256, 0, s>>>(xt, (unsigned long long)xtBatch, F_, w.hx, nullptr, w.h0, totP);
    run_mra(s, w, gate, w.gates, 1);     // sigmoid gates (r|u)
    pack_kernel<<<totP / 256, 256, 0, s>>>(xt, (unsigned long long)xtBatch, F_, w.hx, w.gates, w.h0, totP);
    run_mra(s, w, upd, w.cbuf, 2);       // tanh candidate
    gru_kernel<<<totH / 256, 256, 0, s>>>(w.gates, w.cbuf, w.hx, w.hx16, totH);
}

} // namespace

extern "C" void kernel_launch(void* const* d_in, const int* in_sizes, int n_in,
                              void* d_out, int out_size, void* d_ws, size_t ws_size,
                              hipStream_t stream)
{
    (void)in_sizes; (void)out_size; (void)ws_size;
    if (n_in < 36) return;

    // inputs in setup_inputs() insertion order (params recursively)
    const float* x        = (const float*)d_in[0];
    const float* adj_node = (const float*)d_in[1];
    const float* adj_edge = (const float*)d_in[2];
    const float* Mmat     = (const float*)d_in[3];
    const float* Wi       = (const float*)d_in[4];
    const float* bi       = (const float*)d_in[5];
    const float* raw[4][6];               // {Wn, an, We, ae, Wo, bo} x {eg, eu, dg, du}
    int base[4] = {6, 12, 20, 26};
    for (int m = 0; m < 4; ++m)
        for (int j = 0; j < 6; ++j) raw[m][j] = (const float*)d_in[base[m] + j];
    const float* W_enc = (const float*)d_in[18];
    const float* b_enc = (const float*)d_in[19];
    const float* W_dec = (const float*)d_in[32];
    const float* b_dec = (const float*)d_in[33];
    const float* Wout  = (const float*)d_in[34];
    const float* bout  = (const float*)d_in[35];
    float* out = (float*)d_out;

    // bump allocator over d_ws (256B aligned)
    char* wsbase = (char*)d_ws;
    size_t off = 0;
    auto alloc = [&](size_t bytes) -> void* {
        off = (off + 255) & ~(size_t)255;
        void* p = wsbase + off;
        off += bytes;
        return p;
    };

    Ws w;
    w.An16  = (_Float16*)alloc((size_t)NP * NP * 2);
    w.Ae16  = (_Float16*)alloc((size_t)EP * EP * 2);
    w.Mr16  = (_Float16*)alloc((size_t)NP * EP * 2);
    w.McT16 = (_Float16*)alloc((size_t)EP * NP * 2);
    // converted weights (f16)
    _Float16* w16[4][3];                  // {Wn(2x128x64), We(2x128x64), Wo(128xdout)}
    int douts[4] = {2 * F_, F_, 2 * F_, F_};   // eg, eu, dg, du
    for (int m = 0; m < 4; ++m) {
        w16[m][0] = (_Float16*)alloc(2 * DIN * F_ * 2);
        w16[m][1] = (_Float16*)alloc(2 * DIN * F_ * 2);
        w16[m][2] = (_Float16*)alloc((size_t)DIN * douts[m] * 2);
    }
    _Float16* Wenc16 = (_Float16*)alloc(F_ * F_ * 2);
    _Float16* Wdec16 = (_Float16*)alloc(F_ * F_ * 2);

    w.enc_in  = (float*)alloc((size_t)B_ * P_ * N_ * F_ * 4);
    w.hx      = (float*)alloc((size_t)B_ * NP * F_ * 4);
    w.hx16    = (_Float16*)alloc((size_t)B_ * NP * F_ * 2);
    w.enc_out = (float*)alloc((size_t)B_ * NP * F_ * 4);
    w.h0      = (_Float16*)alloc((size_t)B_ * NP * DIN * 2);
    w.hnA     = (_Float16*)alloc((size_t)B_ * NP * DIN * 2);
    w.hnB     = (_Float16*)alloc((size_t)B_ * NP * DIN * 2);
    w.he0     = (_Float16*)alloc((size_t)B_ * EP * DIN * 2);
    w.heA     = (_Float16*)alloc((size_t)B_ * EP * DIN * 2);
    w.heB     = (_Float16*)alloc((size_t)B_ * EP * DIN * 2);
    w.H1n     = (float*)alloc((size_t)B_ * NP * F_ * 4);
    w.H2n     = (float*)alloc((size_t)B_ * NP * F_ * 4);
    w.H1e     = (float*)alloc((size_t)B_ * EP * F_ * 4);
    w.H2e     = (float*)alloc((size_t)B_ * EP * F_ * 4);
    w.cat2    = (_Float16*)alloc((size_t)B_ * NP * DIN * 2);
    w.eout    = (_Float16*)alloc((size_t)B_ * EP * F_ * 2);
    w.gates   = (float*)alloc((size_t)B_ * NP * DIN * 4);
    w.cbuf    = (float*)alloc((size_t)B_ * NP * F_ * 4);
    w.yt      = (float*)alloc((size_t)B_ * NP * F_ * 4);

    // ---- preprocessing ----
    rownorm_kernel<<<NP, 256, 0, stream>>>(adj_node, w.An16, N_, N_, NP);
    rownorm_kernel<<<EP, 256, 0, stream>>>(adj_edge, w.Ae16, E_, E_, EP);
    rownorm_kernel<<<NP, 256, 0, stream>>>(Mmat, w.Mr16, N_, E_, EP);
    colnormT_kernel<<<EP, 256, 0, stream>>>(Mmat, w.McT16, N_, E_, NP);
    for (int m = 0; m < 4; ++m) {
        int nWn = 2 * DIN * F_;
        f32_to_f16_kernel<<<(nWn + 255) / 256, 256, 0, stream>>>(raw[m][0], w16[m][0], nWn);
        f32_to_f16_kernel<<<(nWn + 255) / 256, 256, 0, stream>>>(raw[m][2], w16[m][1], nWn);
        int nWo = DIN * douts[m];
        f32_to_f16_kernel<<<(nWo + 255) / 256, 256, 0, stream>>>(raw[m][4], w16[m][2], nWo);
    }
    f32_to_f16_kernel<<<(F_ * F_ + 255) / 256, 256, 0, stream>>>(W_enc, Wenc16, F_ * F_);
    f32_to_f16_kernel<<<(F_ * F_ + 255) / 256, 256, 0, stream>>>(W_dec, Wdec16, F_ * F_);
    {
        int tot = B_ * P_ * N_ * F_;
        enc_in_kernel<<<(tot + 255) / 256, 256, 0, stream>>>(x, Wi, bi, w.enc_in, tot);
    }
    hipMemsetAsync(w.hx, 0, (size_t)B_ * NP * F_ * 4, stream);
    hipMemsetAsync(w.hx16, 0, (size_t)B_ * NP * F_ * 2, stream);

    MraP mp[4];
    for (int m = 0; m < 4; ++m) {
        mp[m].Wn0 = w16[m][0];           mp[m].Wn1 = w16[m][0] + DIN * F_;
        mp[m].We0 = w16[m][1];           mp[m].We1 = w16[m][1] + DIN * F_;
        mp[m].Wo  = w16[m][2];
        mp[m].an  = raw[m][1];  mp[m].ae = raw[m][3];  mp[m].bo = raw[m][5];
        mp[m].dout = douts[m];
    }
    const size_t NPF = (size_t)NP * F_;

    // ---- encoder: P steps ----
    for (int p = 0; p < P_; ++p) {
        const float* xt = w.enc_in + (size_t)p * N_ * F_;       // batch stride P*N*F
        run_cell(stream, w, xt, (size_t)P_ * N_ * F_, mp[0], mp[1]);
    }
    // enc_out = sigmoid(hT @ W_enc + b_enc)
    gemm(stream, w.hx16, NPF, F_, Wenc16, 0, F_, w.enc_out, NPF, F_,
         nullptr, 0, 0, b_enc, NP, F_, F_, 1);

    // ---- decoder: T steps ----
    for (int t = 0; t < T_; ++t) {
        run_cell(stream, w, w.enc_out, NPF, mp[2], mp[3]);
        gemm(stream, w.hx16, NPF, F_, Wdec16, 0, F_, w.yt, NPF, F_,
             nullptr, 0, 0, b_dec, NP, F_, F_, 1);
        final_out_kernel<<<(B_ * N_ + 255) / 256, 256, 0, stream>>>(w.yt, Wout, bout, out, t);
    }
}
